// MultiHeadSelfAttention_2430951490280
// MI455X (gfx1250) — matrix operand order repaired and checked by reading, not compiled, not run on hardware
//
#include <hip/hip_runtime.h>

// MI455X / gfx1250, wave32. Causal multi-head self-attention:
//  prep (x->bf16, W->W^T bf16) -> QKV proj (bf16 WMMA, async-LDS staging)
//  -> flash attention (bf16 WMMA, f32 softmax) -> out proj.
// B=2, S=2048, D=1024, H=16, HD=64.

typedef __attribute__((ext_vector_type(16))) __bf16 v16bf;
typedef __attribute__((ext_vector_type(8)))  __bf16 v8bf;
typedef __attribute__((ext_vector_type(4)))  __bf16 v4bf;
typedef __attribute__((ext_vector_type(8)))  float  v8f;
typedef int v4i_g __attribute__((vector_size(16)));   // matches builtin param type

#define SEQ 2048
#define DIM 1024
#define NH  16
#define HD  64
#define BATCH 2
#define KP  40   // padded LDS row stride (bf16) for 32-wide tiles (attention P)
#define KP2 72   // padded LDS row stride (bf16) for 64-wide tiles (GEMM)

#if defined(__has_builtin)
#if __has_builtin(__builtin_amdgcn_global_load_async_to_lds_b128) && \
    __has_builtin(__builtin_amdgcn_s_wait_asynccnt)
#define HAVE_ASYNC 1
#endif
#endif
#ifndef HAVE_ASYNC
#define HAVE_ASYNC 0
#endif

#if HAVE_ASYNC
typedef __attribute__((address_space(1))) v4i_g* gb128_t;
typedef __attribute__((address_space(3))) v4i_g* lb128_t;
#endif

static __device__ __forceinline__ v16bf load16(const __bf16* p0, const __bf16* p1) {
  v8bf lo = *(const v8bf*)p0;
  v8bf hi = *(const v8bf*)p1;
  return __builtin_shufflevector(lo, hi, 0,1,2,3,4,5,6,7,8,9,10,11,12,13,14,15);
}

static __device__ __forceinline__ v8f wmma_bf16(v16bf a, v16bf b, v8f c) {
  return __builtin_amdgcn_wmma_f32_16x16x32_bf16(false, a, false, b, (short)0, c, false, false);
}

static __device__ __forceinline__ float rmax16(float v) {
  #pragma unroll
  for (int m = 1; m < 16; m <<= 1) v = fmaxf(v, __shfl_xor(v, m, 32));
  return v;
}
static __device__ __forceinline__ float rsum16(float v) {
  #pragma unroll
  for (int m = 1; m < 16; m <<= 1) v += __shfl_xor(v, m, 32);
  return v;
}

// Copy a [128 rows][64 bf16] tile (global row stride DIM) into LDS rows of
// stride KP2, as 4 x 16-byte segments per thread (256 threads).
static __device__ __forceinline__ void stage_tile(const __bf16* __restrict__ gbase,
                                                  __bf16* lbase, int tid) {
  #pragma unroll
  for (int i = 0; i < 4; i++) {
    int idx = i * 256 + tid;
    int row = idx >> 3, c8 = (idx & 7) * 8;
    const __bf16* g = gbase + (size_t)row * DIM + c8;
    __bf16* l = lbase + row * KP2 + c8;
#if HAVE_ASYNC
    __builtin_amdgcn_global_load_async_to_lds_b128((gb128_t)g, (lb128_t)l, 0, 0);
#else
    *(v8bf*)l = *(const v8bf*)g;
#endif
  }
}

template <int N>
static __device__ __forceinline__ void wait_async() {
#if HAVE_ASYNC
  __builtin_amdgcn_s_wait_asynccnt(N);
#endif
}

// ---------------------------------------------------------------------------
// Prep kernels: one-time fp32 -> bf16 conversion (x) and transposed bf16
// weights Wt[n][k] = W[k][n], so GEMM staging is a raw row copy.
// ---------------------------------------------------------------------------
__global__ __launch_bounds__(256) void cvt_f32_bf16(const float* __restrict__ src,
                                                    __bf16* __restrict__ dst, int n4) {
  int i = blockIdx.x * 256 + threadIdx.x;   // float4 units
  if (i < n4) {
    float4 v = ((const float4*)src)[i];
    v4bf b = { (__bf16)v.x, (__bf16)v.y, (__bf16)v.z, (__bf16)v.w };
    ((v4bf*)dst)[i] = b;
  }
}

__global__ __launch_bounds__(256) void wtrans(const float* __restrict__ W,
                                              __bf16* __restrict__ Wt) {
  __shared__ float t[32][33];
  const int k0 = blockIdx.x * 32, n0 = blockIdx.y * 32;
  const int tid = threadIdx.x;
  const int r = tid >> 3, c4 = (tid & 7) * 4;
  float4 v = *(const float4*)&W[(size_t)(k0 + r) * DIM + n0 + c4];
  t[r][c4 + 0] = v.x; t[r][c4 + 1] = v.y; t[r][c4 + 2] = v.z; t[r][c4 + 3] = v.w;
  __syncthreads();
  v4bf b = { (__bf16)t[c4 + 0][r], (__bf16)t[c4 + 1][r],
             (__bf16)t[c4 + 2][r], (__bf16)t[c4 + 3][r] };
  *(v4bf*)&Wt[(size_t)(n0 + r) * DIM + k0 + c4] = b;
}

// ---------------------------------------------------------------------------
// 128x128-tile GEMM: C = A(4096xDIM) * Wt^T + bias, K-step 64, double-buffered
// async-LDS staging. A and Wt are bf16, Wt is [N][K].
// MODE 0: dst bf16 head-major [B,H,S,HD]            (Q, K)
// MODE 2: dst bf16 head-major transposed [B,H,HD,S] (V^T)
// MODE 3: dst f32 [B,S,DIM]                          (final output)
// ---------------------------------------------------------------------------
template <int MODE>
__global__ __launch_bounds__(256) void gemm128(
    const __bf16* __restrict__ A, const __bf16* __restrict__ Wt,
    const float* __restrict__ bias,
    __bf16* __restrict__ dstB, float* __restrict__ dstF)
{
  __shared__ __bf16 sA[2][128 * KP2];
  __shared__ __bf16 sB[2][128 * KP2];

  const int tid   = threadIdx.x;
  const int wave  = tid >> 5, lane = tid & 31;
  const int laneN = lane & 15, hh = lane >> 4;
  const int waveM = wave >> 1, waveN = wave & 1;
  const int m0 = blockIdx.x * 128, n0 = blockIdx.y * 128;

  const __bf16* Abase = A  + (size_t)m0 * DIM;   // rows m, k contiguous
  const __bf16* Bbase = Wt + (size_t)n0 * DIM;   // rows n, k contiguous

  const v8f vz = {0.f,0.f,0.f,0.f,0.f,0.f,0.f,0.f};
  v8f acc[2][4];
  #pragma unroll
  for (int i = 0; i < 2; i++)
    #pragma unroll
    for (int j = 0; j < 4; j++) acc[i][j] = vz;

  // Preload first K tile into buffer 0.
  stage_tile(Abase, &sA[0][0], tid);
  stage_tile(Bbase, &sB[0][0], tid);

  for (int k0 = 0; k0 < DIM; k0 += 64) {
    const int cur = (k0 >> 6) & 1, nxt = cur ^ 1;
    const bool more = (k0 + 64) < DIM;
    if (more) {            // stage next tile while computing current
      stage_tile(Abase + k0 + 64, &sA[nxt][0], tid);
      stage_tile(Bbase + k0 + 64, &sB[nxt][0], tid);
      wait_async<8>();     // current tile's 8 older copies complete
    } else {
      wait_async<0>();
    }
    __syncthreads();            // visible to all waves

    #pragma unroll
    for (int kc = 0; kc < 2; kc++) {
      v16bf af[2], bfr[4];
      #pragma unroll
      for (int i = 0; i < 2; i++) {
        const __bf16* p = &sA[cur][(waveM * 32 + i * 16 + laneN) * KP2 + kc * 32 + hh * 16];
        af[i] = load16(p, p + 8);           // K = h*16..+15 contiguous
      }
      #pragma unroll
      for (int j = 0; j < 4; j++) {
        const __bf16* p = &sB[cur][(waveN * 64 + j * 16 + laneN) * KP2 + kc * 32 + hh * 16];
        bfr[j] = load16(p, p + 8);          // K = h*16..+15 contiguous
      }
      #pragma unroll
      for (int i = 0; i < 2; i++)
        #pragma unroll
        for (int j = 0; j < 4; j++)
          acc[i][j] = wmma_bf16(af[i], bfr[j], acc[i][j]);
    }
    __syncthreads();            // done reading cur before it is restaged
  }

  // Epilogue: C-fragment -> lane (hh,laneN), VGPR r: M = r + 8*hh, N = laneN.
  #pragma unroll
  for (int i = 0; i < 2; i++) {
    int mbase = m0 + waveM * 32 + i * 16 + 8 * hh;
    #pragma unroll
    for (int sub = 0; sub < 4; sub++) {
      int n = n0 + waveN * 64 + sub * 16 + laneN;
      float bv = bias[n];
      #pragma unroll
      for (int r = 0; r < 8; r++) {
        int m = mbase + r;
        float cv = acc[i][sub][r] + bv;
        if constexpr (MODE == 3) {
          dstF[(size_t)m * DIM + n] = cv;
        } else {
          int b = m >> 11, s = m & (SEQ - 1);
          int hd = n >> 6, d = n & (HD - 1);
          if constexpr (MODE == 2)
            dstB[(((size_t)b * NH + hd) * HD + d) * SEQ + s] = (__bf16)cv;   // V^T
          else
            dstB[(((size_t)b * NH + hd) * SEQ + s) * HD + d] = (__bf16)cv;   // Q, K
        }
      }
    }
  }
}

// ---------------------------------------------------------------------------
// Flash attention, causal. One wave per 16-row query tile; kv chunks of 32.
// q,k: [B,H,S,HD] bf16; vt: [B,H,HD,S] bf16; out: [B,S,DIM] bf16.
// ---------------------------------------------------------------------------
__global__ __launch_bounds__(256) void attn_kernel(
    const __bf16* __restrict__ qw, const __bf16* __restrict__ kw,
    const __bf16* __restrict__ vtw, __bf16* __restrict__ aout)
{
  __shared__ __bf16 sP[8 * 16 * KP];   // per-wave 16x32 P scratch (padded)

  const int tid = threadIdx.x, wave = tid >> 5, lane = tid & 31;
  const int laneN = lane & 15, hh = lane >> 4;
  const int hi = blockIdx.y, bi = blockIdx.z;
  const int qt = blockIdx.x * 8 + wave;
  const int q0 = qt * 16;

  const size_t headoff = ((size_t)bi * NH + hi) * SEQ * HD;
  const __bf16* qh = qw + headoff;
  const __bf16* kh = kw + headoff;
  const __bf16* vh = vtw + headoff;          // [HD][SEQ]
  __bf16* pw = &sP[wave * 16 * KP];

  // Q as A-operand: two K=32 fragments covering d=0..31, 32..63.
  v16bf qf[2];
  {
    const __bf16* qp = qh + (size_t)(q0 + laneN) * HD;
    #pragma unroll
    for (int kd = 0; kd < 2; kd++) {
      const __bf16* p = qp + kd * 32 + hh * 16;
      qf[kd] = load16(p, p + 8);
    }
  }

  const v8f vz = {0.f,0.f,0.f,0.f,0.f,0.f,0.f,0.f};
  v8f o[4];
  #pragma unroll
  for (int s = 0; s < 4; s++) o[s] = vz;
  float mrow[8], lrow[8];
  #pragma unroll
  for (int r = 0; r < 8; r++) { mrow[r] = -3.0e38f; lrow[r] = 0.f; }

  const int nch = (qt >> 1) + 1;
  for (int c = 0; c < nch; c++) {
    const int kv0 = c * 32;

    // S = Q * K^T : K rows [s_kv][d] are directly the B-operand layout.
    v8f s0 = vz, s1 = vz;
    #pragma unroll
    for (int t = 0; t < 2; t++) {
      const __bf16* kp = kh + (size_t)(kv0 + t * 16 + laneN) * HD;
      v16bf kf0 = load16(kp + hh * 16,      kp + hh * 16 + 8);        // d 0..31
      v16bf kf1 = load16(kp + 32 + hh * 16, kp + 32 + hh * 16 + 8);   // d 32..63
      if (t == 0) { s0 = wmma_bf16(qf[0], kf0, s0); s0 = wmma_bf16(qf[1], kf1, s0); }
      else        { s1 = wmma_bf16(qf[0], kf0, s1); s1 = wmma_bf16(qf[1], kf1, s1); }
    }

    const bool needmask = (kv0 + 31) > q0;   // only the diagonal chunk
    #pragma unroll
    for (int r = 0; r < 8; r++) {
      const int M = r + 8 * hh;
      const int sq = q0 + M;
      float e0 = s0[r] * 0.125f;             // 1/sqrt(64)
      float e1 = s1[r] * 0.125f;
      if (needmask) {
        if (kv0 + laneN      > sq) e0 = -3.0e38f;
        if (kv0 + 16 + laneN > sq) e1 = -3.0e38f;
      }
      float mx = rmax16(fmaxf(e0, e1));
      float mnew = fmaxf(mrow[r], mx);
      float alpha = __expf(mrow[r] - mnew);
      float p0 = __expf(e0 - mnew);
      float p1 = __expf(e1 - mnew);
      lrow[r] = lrow[r] * alpha + rsum16(p0 + p1);
      mrow[r] = mnew;
      #pragma unroll
      for (int sub = 0; sub < 4; sub++) o[sub][r] *= alpha;
      // C-layout -> LDS [M][kv 0..31]
      pw[M * KP + laneN]      = (__bf16)p0;
      pw[M * KP + 16 + laneN] = (__bf16)p1;
    }

    // Reload P in A-fragment layout, multiply by V (B-operand = V^T rows).
    const __bf16* pp = pw + laneN * KP + hh * 16;
    v16bf pf = load16(pp, pp + 8);
    #pragma unroll
    for (int sub = 0; sub < 4; sub++) {
      const __bf16* vp = vh + (size_t)(sub * 16 + laneN) * SEQ + kv0 + hh * 16;
      v16bf vf = load16(vp, vp + 8);
      o[sub] = wmma_bf16(pf, vf, o[sub]);
    }
  }

  // Normalize and store attn output as bf16 [B,S,DIM] (input to out-proj).
  #pragma unroll
  for (int r = 0; r < 8; r++) {
    const int sq = q0 + r + 8 * hh;
    const float inv = 1.0f / lrow[r];
    #pragma unroll
    for (int sub = 0; sub < 4; sub++) {
      int d = sub * 16 + laneN;
      aout[((size_t)bi * SEQ + sq) * DIM + hi * HD + d] = (__bf16)(o[sub][r] * inv);
    }
  }
}

// ---------------------------------------------------------------------------
extern "C" void kernel_launch(void* const* d_in, const int* in_sizes, int n_in,
                              void* d_out, int out_size, void* d_ws, size_t ws_size,
                              hipStream_t stream) {
  (void)in_sizes; (void)n_in; (void)out_size; (void)ws_size;
  const float* x  = (const float*)d_in[0];
  const float* Wq = (const float*)d_in[1];
  const float* bq = (const float*)d_in[2];
  const float* Wk = (const float*)d_in[3];
  const float* bk = (const float*)d_in[4];
  const float* Wv = (const float*)d_in[5];
  const float* bv = (const float*)d_in[6];
  const float* Wo = (const float*)d_in[7];
  const float* bo = (const float*)d_in[8];

  const size_t SEG = (size_t)BATCH * NH * SEQ * HD * sizeof(__bf16); // 8 MB
  const size_t WSEG = (size_t)DIM * DIM * sizeof(__bf16);            // 2 MB
  char* ws = (char*)d_ws;
  __bf16* qw  = (__bf16*)(ws + 0 * SEG);
  __bf16* kw  = (__bf16*)(ws + 1 * SEG);
  __bf16* vtw = (__bf16*)(ws + 2 * SEG);
  __bf16* aw  = (__bf16*)(ws + 3 * SEG);
  __bf16* xb  = (__bf16*)(ws + 4 * SEG);
  __bf16* wtq = (__bf16*)(ws + 5 * SEG + 0 * WSEG);
  __bf16* wtk = (__bf16*)(ws + 5 * SEG + 1 * WSEG);
  __bf16* wtv = (__bf16*)(ws + 5 * SEG + 2 * WSEG);
  __bf16* wto = (__bf16*)(ws + 5 * SEG + 3 * WSEG);

  // Prep: x -> bf16, weights -> transposed bf16.
  const int n4 = (BATCH * SEQ * DIM) / 4;
  cvt_f32_bf16<<<dim3((n4 + 255) / 256), 256, 0, stream>>>(x, xb, n4);
  dim3 tg(DIM / 32, DIM / 32);
  wtrans<<<tg, 256, 0, stream>>>(Wq, wtq);
  wtrans<<<tg, 256, 0, stream>>>(Wk, wtk);
  wtrans<<<tg, 256, 0, stream>>>(Wv, wtv);
  wtrans<<<tg, 256, 0, stream>>>(Wo, wto);

  dim3 gg(32, 8), bb(256);
  gemm128<0><<<gg, bb, 0, stream>>>(xb, wtq, bq, qw,  nullptr);
  gemm128<0><<<gg, bb, 0, stream>>>(xb, wtk, bk, kw,  nullptr);
  gemm128<2><<<gg, bb, 0, stream>>>(xb, wtv, bv, vtw, nullptr);

  attn_kernel<<<dim3(SEQ / 16 / 8, NH, BATCH), 256, 0, stream>>>(qw, kw, vtw, aw);

  gemm128<3><<<gg, bb, 0, stream>>>(aw, wto, bo, nullptr, (float*)d_out);
}
